// LSTMCell_7988639171179
// MI455X (gfx1250) — matrix operand order repaired and checked by reading, not compiled, not run on hardware
//
#include <hip/hip_runtime.h>

#define HS   128      // hidden size
#define KDIM 256      // input + hidden
#define NDIM 512      // 4 * HS (f,i,o,c gates)
#define BM   128      // batch rows per workgroup (8 waves x 16 rows)
#define RPB  264      // LDS row pitch in bf16 elems (528 B = 132 dwords; 132 % 64 = 4 banks shift/row)

typedef __attribute__((ext_vector_type(16))) __bf16 v16bf;
typedef __attribute__((ext_vector_type(8)))  __bf16 v8bf;
typedef __attribute__((ext_vector_type(8)))  float  v8f;
typedef __attribute__((ext_vector_type(4)))  float  f32x4;
typedef __attribute__((ext_vector_type(4)))  unsigned int u32x4;
typedef __attribute__((ext_vector_type(8)))  int    i32x8;
typedef __attribute__((ext_vector_type(4)))  int    i32x4;

union BF16x16 { v16bf v; u32x4 u[2]; };
union BF16x8  { v8bf  v; u32x4 u;    };

__device__ __forceinline__ float fast_sigmoid(float x) {
  return 1.0f / (1.0f + __expf(-x));
}
__device__ __forceinline__ float fast_tanh(float x) {
  float e = __expf(-2.0f * fabsf(x));
  return copysignf((1.0f - e) / (1.0f + e), x);
}

extern __shared__ unsigned char lds_raw[];

// ---- One-shot pack: 4x f32 [128,256] gate weights -> bf16 [512,256] in d_ws ----
__global__ void __launch_bounds__(256)
lstm_pack_w(const float* __restrict__ Wf, const float* __restrict__ Wi,
            const float* __restrict__ Wo, const float* __restrict__ Wc,
            unsigned short* __restrict__ Wp)
{
  const int j = blockIdx.x * 256 + threadIdx.x;   // 16384 chunks of 8 floats
  const int n = j >> 5;                           // packed gate row 0..511
  const int c = (j & 31) << 3;                    // col 0..248
  const float* src = (n < 256) ? ((n < 128) ? Wf : Wi)
                               : ((n < 384) ? Wo : Wc);
  const float* p = src + (n & 127) * KDIM + c;
  const f32x4 f0 = *(const f32x4*)(p);
  const f32x4 f1 = *(const f32x4*)(p + 4);
  BF16x8 t;
  t.v[0] = (__bf16)f0[0]; t.v[1] = (__bf16)f0[1];
  t.v[2] = (__bf16)f0[2]; t.v[3] = (__bf16)f0[3];
  t.v[4] = (__bf16)f1[0]; t.v[5] = (__bf16)f1[1];
  t.v[6] = (__bf16)f1[2]; t.v[7] = (__bf16)f1[3];
  *(u32x4*)((__bf16*)Wp + n * KDIM + c) = t.u;
}

__global__ void __launch_bounds__(256)
lstm_cell_wmma(const float* __restrict__ inp, const float* __restrict__ hid,
               const float* __restrict__ cell,
               const float* __restrict__ Wf, const float* __restrict__ Wi,
               const float* __restrict__ Wo, const float* __restrict__ Wc,
               const unsigned short* __restrict__ Wp, int use_tdm,
               const float* __restrict__ bfp, const float* __restrict__ bip,
               const float* __restrict__ bop, const float* __restrict__ bcp,
               float* __restrict__ out_h, float* __restrict__ out_c)
{
  __bf16* ldsW = (__bf16*)lds_raw;
  const int tid = threadIdx.x;

  if (use_tdm) {
    // ---- Stage W via Tensor Data Mover: one DMA, padded rows done in HW ----
    // D# (CDNA5 ISA): 2D tensor, data_size=2B,
    // tensor_dim0=256, tensor_dim1=512, tile=[256,512], stride0=256,
    // pad_enable=1, pad_interval=6 (128 dwords = one 512B row),
    // pad_amount=3 (4 dwords = 16B) -> LDS pitch 528B == RPB.
    if (tid < 32) {            // wave 0 issues the DMA (TDM ignores EXEC)
      const unsigned long long ga = (unsigned long long)(uintptr_t)Wp;
      u32x4 g0 = { 1u,                                    // count=1, user desc
                   0u,                                    // lds_addr = 0
                   (unsigned)(ga & 0xFFFFFFFFu),          // global_addr[31:0]
                   (unsigned)(((ga >> 32) & 0x1FFFFFFu) | (2u << 30)) }; // [56:32] | type=2
      i32x8 g1 = { (int)((1u << 16) |                     // data_size = 2B
                         (1u << 20) |                     // pad_enable
                         (6u << 22) |                     // pad_interval: 128 dwords
                         (3u << 25)),                     // pad_amount: 4 dwords
                   (int)(256u << 16),                     // tensor_dim0[15:0] @ bits 63:48
                   (int)(512u << 16),                     // tensor_dim1[15:0] @ bits 95:80
                   (int)(256u << 16),                     // tile_dim0 @ bits 127:112
                   512,                                   // tile_dim1
                   256,                                   // tensor_dim0_stride[31:0]
                   0, 0 };
      i32x4 g2 = {0, 0, 0, 0};
      i32x4 g3 = {0, 0, 0, 0};
      i32x8 g4 = {0, 0, 0, 0, 0, 0, 0, 0};                // unused higher-dim group (2D tensor)
      __builtin_amdgcn_tensor_load_to_lds(g0, g1, g2, g3, g4, 0);
      __builtin_amdgcn_s_wait_tensorcnt(0);
    }
    __syncthreads();
  } else {
    // ---- Fallback: cooperative VALU staging with on-the-fly f32->bf16 ----
    #pragma unroll 4
    for (int i = 0; i < 64; ++i) {
      const int j = tid + (i << 8);
      const int n = j >> 5;              // 0..511
      const int c = (j & 31) << 3;       // 0..248
      const float* src = (n < 256) ? ((n < 128) ? Wf : Wi)
                                   : ((n < 384) ? Wo : Wc);
      const float* p = src + (n & 127) * KDIM + c;
      const f32x4 f0 = *(const f32x4*)(p);
      const f32x4 f1 = *(const f32x4*)(p + 4);
      BF16x8 t;
      t.v[0] = (__bf16)f0[0]; t.v[1] = (__bf16)f0[1];
      t.v[2] = (__bf16)f0[2]; t.v[3] = (__bf16)f0[3];
      t.v[4] = (__bf16)f1[0]; t.v[5] = (__bf16)f1[1];
      t.v[6] = (__bf16)f1[2]; t.v[7] = (__bf16)f1[3];
      *(u32x4*)(ldsW + n * RPB + c) = t.u;
    }
    __syncthreads();
  }

  const int lane = tid & 31;
  const int wave = tid >> 5;
  const int mrow = lane & 15;          // row-within-tile (A) / column (B,C,D)
  const int hi   = lane >> 4;          // lane-half select
  const int m_base = blockIdx.x * BM + wave * 16;

  // ---- Build A fragments for this wave's 16 batch rows: 8 k-steps of 32 ----
  // 16-bit A layout (ISA 7.12.2): lanes 0-15 hold K k0+0..7 (v0-3) and k0+16..23 (v4-7);
  // lanes 16-31 hold k0+8..15 and k0+24..31. combined[:,0:128]=input, [:,128:256]=hidden.
  v16bf A[8];
  {
    const int row = m_base + mrow;
    const float* pin = inp + (size_t)row * HS;
    const float* phd = hid + (size_t)row * HS;
    #pragma unroll
    for (int kk = 0; kk < 8; ++kk) {
      const float* src = (kk < 4) ? pin : phd;
      const int kb = (kk & 3) * 32 + hi * 8;
      const f32x4 f0 = *(const f32x4*)(src + kb);
      const f32x4 f1 = *(const f32x4*)(src + kb + 4);
      const f32x4 f2 = *(const f32x4*)(src + kb + 16);
      const f32x4 f3 = *(const f32x4*)(src + kb + 20);
      v16bf a;
      a[0]  = (__bf16)f0[0]; a[1]  = (__bf16)f0[1]; a[2]  = (__bf16)f0[2]; a[3]  = (__bf16)f0[3];
      a[4]  = (__bf16)f1[0]; a[5]  = (__bf16)f1[1]; a[6]  = (__bf16)f1[2]; a[7]  = (__bf16)f1[3];
      a[8]  = (__bf16)f2[0]; a[9]  = (__bf16)f2[1]; a[10] = (__bf16)f2[2]; a[11] = (__bf16)f2[3];
      a[12] = (__bf16)f3[0]; a[13] = (__bf16)f3[1]; a[14] = (__bf16)f3[2]; a[15] = (__bf16)f3[3];
      A[kk] = a;
    }
  }

  // B fragment (32x16 bf16): lane L holds K of column n0+L%16 in A's order
  // (lanes 0-15: K=k0..k0+7,k0+16..k0+23; lanes 16-31: K=k0+8..k0+15,k0+24..k0+31).
  // Base pointer per gate; k0 is a compile-time constant in the unrolled k-loop
  // so both halves land in the 16-bit DS immediate offset (max 480 B).
  auto loadB = [](const __bf16* p, int k0) -> v16bf {
    BF16x16 t;
    t.u[0] = *(const u32x4*)(p + k0);
    t.u[1] = *(const u32x4*)(p + k0 + 16);
    return t.v;
  };

  // ---- Main loop: 8 h-tiles x (8 k-steps x 4 gate WMMAs) ----
  for (int t = 0; t < 8; ++t) {
    const int h0 = t * 16;
    // One LDS base address register per gate (gates are 128*RPB*2 = 67584 B
    // apart -- beyond the DS immediate range, so they must be registers).
    const __bf16* pF = ldsW + (h0 + mrow) * RPB + hi * 8;
    const __bf16* pI = pF + 128 * RPB;
    const __bf16* pO = pF + 256 * RPB;
    const __bf16* pC = pF + 384 * RPB;

    v8f accF = {0,0,0,0,0,0,0,0};
    v8f accI = {0,0,0,0,0,0,0,0};
    v8f accO = {0,0,0,0,0,0,0,0};
    v8f accC = {0,0,0,0,0,0,0,0};
    #pragma unroll
    for (int kk = 0; kk < 8; ++kk) {
      const int k0 = kk * 32;
      const v16bf Bf = loadB(pF, k0);
      const v16bf Bi = loadB(pI, k0);
      const v16bf Bo = loadB(pO, k0);
      const v16bf Bc = loadB(pC, k0);
      accF = __builtin_amdgcn_wmma_f32_16x16x32_bf16(false, A[kk], false, Bf, (short)0, accF, false, false);
      accI = __builtin_amdgcn_wmma_f32_16x16x32_bf16(false, A[kk], false, Bi, (short)0, accI, false, false);
      accO = __builtin_amdgcn_wmma_f32_16x16x32_bf16(false, A[kk], false, Bo, (short)0, accO, false, false);
      accC = __builtin_amdgcn_wmma_f32_16x16x32_bf16(false, A[kk], false, Bc, (short)0, accC, false, false);
    }

    // ---- Epilogue: all 4 gates for (b, h) live in the same lane/VGPR ----
    // C/D layout: VGPR r, lanes 0-15 -> M=r, N=lane; lanes 16-31 -> M=r+8, N=lane-16.
    const int h = h0 + mrow;
    const float bF = bfp[h], bI = bip[h], bO = bop[h], bC = bcp[h];
    #pragma unroll
    for (int r = 0; r < 8; ++r) {
      const int m = m_base + r + hi * 8;
      const size_t idx = (size_t)m * HS + h;
      const float fg = fast_sigmoid(accF[r] + bF);
      const float ig = fast_sigmoid(accI[r] + bI);
      const float og = fast_sigmoid(accO[r] + bO);
      const float gg = fast_tanh(accC[r] + bC);
      const float cN = fg * cell[idx] + ig * gg;
      out_c[idx] = cN;
      out_h[idx] = og * fast_tanh(cN);
    }
  }
}

extern "C" void kernel_launch(void* const* d_in, const int* in_sizes, int n_in,
                              void* d_out, int out_size, void* d_ws, size_t ws_size,
                              hipStream_t stream) {
  (void)n_in; (void)out_size;
  const float* inp = (const float*)d_in[0];
  const float* hid = (const float*)d_in[1];
  const float* cel = (const float*)d_in[2];
  const float* Wf  = (const float*)d_in[3];
  const float* bf  = (const float*)d_in[4];
  const float* Wi  = (const float*)d_in[5];
  const float* bi  = (const float*)d_in[6];
  const float* Wo  = (const float*)d_in[7];
  const float* bo  = (const float*)d_in[8];
  const float* Wc  = (const float*)d_in[9];
  const float* bc  = (const float*)d_in[10];

  const int batch = in_sizes[0] / HS;          // 65536
  float* out_h = (float*)d_out;
  float* out_c = out_h + (size_t)batch * HS;

  // Use TDM path only if scratch can hold the packed bf16 W (256 KB).
  const size_t packedW = (size_t)NDIM * KDIM * sizeof(unsigned short);
  const int use_tdm = (d_ws != nullptr && ws_size >= packedW) ? 1 : 0;
  unsigned short* Wp = (unsigned short*)d_ws;

  if (use_tdm) {
    lstm_pack_w<<<64, 256, 0, stream>>>(Wf, Wi, Wo, Wc, Wp);
  }

  const int ldsBytes = NDIM * RPB * 2;         // 270336 B — needs CDNA5's 320KB/WGP LDS
  (void)hipFuncSetAttribute(reinterpret_cast<const void*>(lstm_cell_wmma),
                            hipFuncAttributeMaxDynamicSharedMemorySize, ldsBytes);

  dim3 grid(batch / BM), block(256);
  lstm_cell_wmma<<<grid, block, ldsBytes, stream>>>(inp, hid, cel,
                                                    Wf, Wi, Wo, Wc,
                                                    Wp, use_tdm,
                                                    bf, bi, bo, bc,
                                                    out_h, out_c);
}
